// MultiHeadAttentionWithMask_4355096838648
// MI455X (gfx1250) — matrix operand order repaired and checked by reading, not compiled, not run on hardware
//
#include <hip/hip_runtime.h>
#include <hip/hip_bf16.h>

// ---------------- types ----------------
typedef _Float16 f16;
typedef __attribute__((ext_vector_type(16))) _Float16 v16h;
typedef __attribute__((ext_vector_type(8)))  _Float16 v8h;
typedef __attribute__((ext_vector_type(4)))  _Float16 v4h;
typedef __attribute__((ext_vector_type(8)))  float    v8f;
typedef __attribute__((ext_vector_type(4)))  float    v4f;

#define S_TOK 64
#define DMODEL 256
#define NHEAD 4
#define HD 64
#define CT_STRIDE ((size_t)8 * 32 * 16)        // halves per column-tile strip
#define MAT_STRIDE ((size_t)16 * CT_STRIDE)    // halves per matrix in wfrag

// WMMA wrapper: D = A*B + C, f16 inputs, f32 accum (gfx1250 shape 16x16x32)
__device__ __forceinline__ v8f wmma_f32_f16(v16h a, v16h b, v8f c) {
    return __builtin_amdgcn_wmma_f32_16x16x32_f16(
        /*neg_a=*/false, a, /*neg_b=*/false, b,
        /*c_mod=*/(short)0, c, /*reuse_a=*/false, /*reuse_b=*/false);
}

// A-fragment (16x32 f16) gather:
// lane L holds row M = L&15; halves 0..15 = K in [base1, base1+16),
// base1 = kc*32 + (L<16 ? 0 : 16).
__device__ __forceinline__ v16h ld_afrag(const f16* base, int row, int stride,
                                         int kc, int lane) {
    int b1 = kc * 32 + ((lane & 16) ? 16 : 0);
    const f16* p = base + row * stride + b1;
    union { v16h v; v8h h[2]; } u;
    u.h[0] = *(const v8h*)(p);
    u.h[1] = *(const v8h*)(p + 8);
    return u.v;
}

// B-fragment: 16 contiguous halves (column n = lane&15 picked by caller;
// K = (lane<16 ? 0 : 16) + j within the 32-K chunk).
__device__ __forceinline__ v16h ld_bfrag(const f16* p) {
    union { v16h v; v8h h[2]; } u;
    u.h[0] = *(const v8h*)(p);
    u.h[1] = *(const v8h*)(p + 8);
    return u.v;
}

// One matrix's worth of column strips for a fixed row tile; all pointers are
// loop-invariant scalars (no per-iteration selects in the hot loop).
__device__ __forceinline__ void gemm_strips(const v16h (&afr)[8],
                                            const f16* __restrict__ wfrag_m,
                                            const float* __restrict__ bias,
                                            f16* __restrict__ outg,
                                            int n, int mt, int half, int lane) {
    int rowadd = (lane & 16) ? 8 : 0;
    for (int ct = half * 8; ct < half * 8 + 8; ++ct) {
        const f16* wf = wfrag_m + (size_t)ct * CT_STRIDE;
        v8f acc = v8f{};
#pragma unroll
        for (int kc = 0; kc < 8; ++kc) {
            v16h b = ld_bfrag(wf + (size_t)(kc * 32 + lane) * 16);
            acc = wmma_f32_f16(afr[kc], b, acc);
        }
        int ncol = ct * 16 + (lane & 15);
        float bb = bias[ncol];
#pragma unroll
        for (int r = 0; r < 8; ++r) {
            int row = mt * 16 + r + rowadd;
            outg[((size_t)n * S_TOK + row) * DMODEL + ncol] = (f16)(acc[r] + bb);
        }
    }
}

// ---------------- kernel 0: repack weights to fp16 B-fragment order --------
// layout: wfrag[((mat*16 + ct)*8 + kc)*32 + lane][16 halves]
__global__ void k_prep_w(const float* __restrict__ wq, const float* __restrict__ wk,
                         const float* __restrict__ wv, const float* __restrict__ wo,
                         f16* __restrict__ wfrag) {
    int idx  = blockIdx.x * blockDim.x + threadIdx.x;   // 0 .. 16383
    int lane = idx & 31;
    int kc   = (idx >> 5) & 7;
    int ct   = (idx >> 8) & 15;
    int m    = idx >> 12;
    const float* W = (m == 0) ? wq : (m == 1) ? wk : (m == 2) ? wv : wo;
    int ncol  = ct * 16 + (lane & 15);
    int kbase = kc * 32 + ((lane & 16) ? 16 : 0);
    f16* dst = wfrag + (size_t)idx * 16;
#pragma unroll
    for (int j = 0; j < 16; ++j)
        dst[j] = (f16)W[(kbase + j) * DMODEL + ncol];
}

// ---------------- kernel 1: LayerNorm + QKV projection ----------------------
// Wave -> fixed row-tile: A-fragments (8) live in registers, B streams from L2.
__global__ void k_ln_qkv(const float* __restrict__ xf,
                         const float* __restrict__ lnw, const float* __restrict__ lnb,
                         const float* __restrict__ bq, const float* __restrict__ bk,
                         const float* __restrict__ bv,
                         const f16* __restrict__ wfrag,
                         f16* __restrict__ qg, f16* __restrict__ kg, f16* __restrict__ vg) {
    int n = blockIdx.x;
    int t = threadIdx.x;                 // 256 threads = 8 waves
    __shared__ float s_lnw[DMODEL];
    __shared__ float s_lnb[DMODEL];
    __shared__ __align__(16) f16 s_xn[S_TOK * DMODEL];

    s_lnw[t] = lnw[t];
    s_lnb[t] = lnb[t];
    __syncthreads();

    if (t < S_TOK) {
        const float* xr = xf + ((size_t)n * S_TOK + t) * DMODEL;
        float mu = 0.f;
#pragma unroll 4
        for (int k = 0; k < DMODEL; ++k) mu += xr[k];
        mu *= (1.f / DMODEL);
        float var = 0.f;
#pragma unroll 4
        for (int k = 0; k < DMODEL; ++k) { float d = xr[k] - mu; var += d * d; }
        var *= (1.f / DMODEL);
        float inv = rsqrtf(var + 1e-5f);
#pragma unroll 4
        for (int k = 0; k < DMODEL; ++k)
            s_xn[t * DMODEL + k] = (f16)((xr[k] - mu) * inv * s_lnw[k] + s_lnb[k]);
    }
    __syncthreads();

    int wave = t >> 5, lane = t & 31;
    int mt   = wave & 3;          // row tile owned by this wave
    int half = wave >> 2;         // column-tile range per matrix: [half*8, half*8+8)

    // preload the 8 A-fragments for this row tile (64 VGPRs, reused 24x)
    v16h afr[8];
#pragma unroll
    for (int kc = 0; kc < 8; ++kc)
        afr[kc] = ld_afrag(s_xn, mt * 16 + (lane & 15), DMODEL, kc, lane);

    // three fixed-matrix passes: pointers stay in SGPRs, no selects in the loop
    gemm_strips(afr, wfrag,                  bq, qg, n, mt, half, lane);
    gemm_strips(afr, wfrag + 1 * MAT_STRIDE, bk, kg, n, mt, half, lane);
    gemm_strips(afr, wfrag + 2 * MAT_STRIDE, bv, vg, n, mt, half, lane);
}

// ---------------- kernel 2: per-edge masked attention + segment-sum ---------
__global__ void k_edge_attn(const f16* __restrict__ qg, const f16* __restrict__ kg,
                            const f16* __restrict__ vg,
                            const float* __restrict__ masks,
                            const float* __restrict__ etw,
                            const int* __restrict__ eidx, const int* __restrict__ etyp,
                            float* __restrict__ ctx, int E) {
    int e = blockIdx.x;
    int t = threadIdx.x;                 // 256 threads = 8 waves
    int wave = t >> 5, lane = t & 31;
    int src = eidx[e];
    int dst = eidx[E + e];
    // per-edge bias: clamp(softplus(etw[type-1]), 1e-6, 1e6)
    float x = etw[etyp[e] - 1];
    float sp = (x > 20.f) ? x : log1pf(__expf(x));
    float ew = fminf(fmaxf(sp, 1e-6f), 1e6f);

    __shared__ float s_km[S_TOK];
    __shared__ float s_qm[S_TOK];
    __shared__ __align__(16) f16 s_q[S_TOK * HD];
    __shared__ __align__(16) f16 s_k[S_TOK * HD];
    __shared__ __align__(16) f16 s_vT[HD * S_TOK];
    __shared__ __align__(16) f16 s_p[S_TOK * S_TOK];
    __shared__ float s_s[S_TOK * S_TOK];

    if (t < S_TOK) {
        s_km[t] = masks[(size_t)dst * S_TOK + t];
        s_qm[t] = masks[(size_t)src * S_TOK + t];
    }

    for (int h = 0; h < NHEAD; ++h) {
        // stage Q_h, K_h (row-major) and V_h^T; 16B vector loads, coalesced
        for (int i = t; i < S_TOK * HD / 8; i += 256) {   // 512 chunks of 8 halves
            int tok = i >> 3;
            int d8  = (i & 7) * 8;
            size_t qb = ((size_t)src * S_TOK + tok) * DMODEL + h * HD + d8;
            size_t kb = ((size_t)dst * S_TOK + tok) * DMODEL + h * HD + d8;
            v8h qv = *(const v8h*)(qg + qb);
            v8h kv = *(const v8h*)(kg + kb);
            v8h vv = *(const v8h*)(vg + kb);
            *(v8h*)(s_q + tok * HD + d8) = qv;
            *(v8h*)(s_k + tok * HD + d8) = kv;
#pragma unroll
            for (int j = 0; j < 8; ++j)
                s_vT[(d8 + j) * S_TOK + tok] = vv[j];
        }
        // prefetch next head's slices into cache while we compute this head
        if (h + 1 < NHEAD) {
            int tok  = t & 63;
            int part = (t >> 6) & 1;
            size_t o = (size_t)tok * DMODEL + (h + 1) * HD + part * 32;
            if (t < 128) {
                __builtin_prefetch(qg + (size_t)src * S_TOK * DMODEL + o, 0, 0);
                __builtin_prefetch(vg + (size_t)dst * S_TOK * DMODEL + o, 0, 0);
            } else {
                __builtin_prefetch(kg + (size_t)dst * S_TOK * DMODEL + o, 0, 0);
            }
        }
        __syncthreads();

        // scores = Q K^T / sqrt(64) + ew, key-masked; 16 tiles, 2 per wave
#pragma unroll
        for (int ti = 0; ti < 2; ++ti) {
            int tile = wave * 2 + ti;
            int qt = tile >> 2, kt = tile & 3;
            v8f acc = v8f{};
#pragma unroll
            for (int kc = 0; kc < 2; ++kc) {
                v16h a = ld_afrag(s_q, qt * 16 + (lane & 15), HD, kc, lane);
                v16h b = ld_bfrag(s_k + (kt * 16 + (lane & 15)) * HD
                                      + kc * 32 + ((lane & 16) ? 16 : 0));
                acc = wmma_f32_f16(a, b, acc);
            }
            int col = kt * 16 + (lane & 15);
            float kmv = s_km[col];
            int rowadd = (lane & 16) ? 8 : 0;
#pragma unroll
            for (int r = 0; r < 8; ++r) {
                float v = acc[r] * 0.125f + ew;       // 1/sqrt(HD)=1/8
                if (!(kmv > 0.f)) v = -1e30f;
                s_s[(qt * 16 + r + rowadd) * S_TOK + col] = v;
            }
        }
        __syncthreads();

        // row softmax over keys, then * km; P stored as fp16
        if (t < S_TOK) {
            float mx = -3.4e38f;
#pragma unroll 4
            for (int c = 0; c < S_TOK; ++c) mx = fmaxf(mx, s_s[t * S_TOK + c]);
            float sum = 0.f;
#pragma unroll 4
            for (int c = 0; c < S_TOK; ++c) {
                float p = __expf(s_s[t * S_TOK + c] - mx);
                s_s[t * S_TOK + c] = p;
                sum += p;
            }
            float inv = 1.f / sum;
#pragma unroll 4
            for (int c = 0; c < S_TOK; ++c)
                s_p[t * S_TOK + c] = (f16)(s_s[t * S_TOK + c] * inv * s_km[c]);
        }
        __syncthreads();

        // ctx = P @ V_h, query-masked, atomically accumulated into ctx[src]
#pragma unroll
        for (int ti = 0; ti < 2; ++ti) {
            int tile = wave * 2 + ti;
            int qt = tile >> 2, nt = tile & 3;
            v8f acc = v8f{};
#pragma unroll
            for (int kc = 0; kc < 2; ++kc) {
                v16h a = ld_afrag(s_p, qt * 16 + (lane & 15), S_TOK, kc, lane);
                v16h b = ld_bfrag(s_vT + (nt * 16 + (lane & 15)) * S_TOK
                                       + kc * 32 + ((lane & 16) ? 16 : 0));
                acc = wmma_f32_f16(a, b, acc);
            }
            int col = h * HD + nt * 16 + (lane & 15);
            int rowadd = (lane & 16) ? 8 : 0;
#pragma unroll
            for (int r = 0; r < 8; ++r) {
                int row = qt * 16 + r + rowadd;
                float v = acc[r] * s_qm[row];
                unsafeAtomicAdd(&ctx[((size_t)src * S_TOK + row) * DMODEL + col], v);
            }
        }
        __syncthreads();
    }
}

// ---------------- kernel 3: output projection + residual + mask -------------
__global__ void k_out(const float* __restrict__ ctx, const f16* __restrict__ wfrag,
                      const float* __restrict__ bo, const float* __restrict__ xf,
                      const float* __restrict__ masks, float* __restrict__ out) {
    int n = blockIdx.x;
    int t = threadIdx.x;
    int wave = t >> 5, lane = t & 31;
    __shared__ __align__(16) f16 s_c[S_TOK * DMODEL];
    __shared__ float s_m[S_TOK];

    if (t < S_TOK) s_m[t] = masks[(size_t)n * S_TOK + t];
    // fp32 -> fp16 staging with 16B loads / 8B stores
    for (int i = t; i < S_TOK * DMODEL / 4; i += 256) {
        v4f c = *(const v4f*)(ctx + (size_t)n * S_TOK * DMODEL + i * 4);
        v4h hh;
#pragma unroll
        for (int j = 0; j < 4; ++j) hh[j] = (f16)c[j];
        *(v4h*)(s_c + i * 4) = hh;
    }
    __syncthreads();

    int mt   = wave & 3;
    int half = wave >> 2;

    v16h afr[8];
#pragma unroll
    for (int kc = 0; kc < 8; ++kc)
        afr[kc] = ld_afrag(s_c, mt * 16 + (lane & 15), DMODEL, kc, lane);

    const f16* wo_frag = wfrag + 3 * MAT_STRIDE;   // matrix index 3
    int rowadd = (lane & 16) ? 8 : 0;
    for (int ct = half * 8; ct < half * 8 + 8; ++ct) {
        const f16* wf = wo_frag + (size_t)ct * CT_STRIDE;
        v8f acc = v8f{};
#pragma unroll
        for (int kc = 0; kc < 8; ++kc) {
            v16h b = ld_bfrag(wf + (size_t)(kc * 32 + lane) * 16);
            acc = wmma_f32_f16(afr[kc], b, acc);
        }
        int ncol = ct * 16 + (lane & 15);
        float bb = bo[ncol];
#pragma unroll
        for (int r = 0; r < 8; ++r) {
            int row = mt * 16 + r + rowadd;
            size_t gi = ((size_t)n * S_TOK + row) * DMODEL + ncol;
            out[gi] = (acc[r] + bb + xf[gi]) * s_m[row];
        }
    }
}

// ---------------- host-side launcher ----------------------------------------
extern "C" void kernel_launch(void* const* d_in, const int* in_sizes, int n_in,
                              void* d_out, int out_size, void* d_ws, size_t ws_size,
                              hipStream_t stream) {
    const float* node_features = (const float*)d_in[0];
    const float* node_masks    = (const float*)d_in[1];
    const float* ln_w          = (const float*)d_in[2];
    const float* ln_b          = (const float*)d_in[3];
    const float* wq            = (const float*)d_in[4];
    const float* bq            = (const float*)d_in[5];
    const float* wk            = (const float*)d_in[6];
    const float* bk            = (const float*)d_in[7];
    const float* wv            = (const float*)d_in[8];
    const float* bv            = (const float*)d_in[9];
    const float* wo            = (const float*)d_in[10];
    const float* bo            = (const float*)d_in[11];
    const float* etw           = (const float*)d_in[12];
    const int*   eidx          = (const int*)d_in[13];
    const int*   etyp          = (const int*)d_in[14];
    float*       out           = (float*)d_out;

    int N = in_sizes[0] / (S_TOK * DMODEL);
    int E = in_sizes[13] / 2;

    // workspace layout
    size_t off = 0;
    f16* wfrag = (f16*)((char*)d_ws + off);           // 4 matrices, fragment order
    off += 4 * MAT_STRIDE * sizeof(f16);
    size_t qkv_bytes = (size_t)N * S_TOK * DMODEL * sizeof(f16);
    f16* qg = (f16*)((char*)d_ws + off); off += qkv_bytes;
    f16* kg = (f16*)((char*)d_ws + off); off += qkv_bytes;
    f16* vg = (f16*)((char*)d_ws + off); off += qkv_bytes;
    float* ctx = (float*)((char*)d_ws + off);
    size_t ctx_bytes = (size_t)N * S_TOK * DMODEL * sizeof(float);

    hipMemsetAsync(ctx, 0, ctx_bytes, stream);

    k_prep_w<<<64, 256, 0, stream>>>(wq, wk, wv, wo, wfrag);
    k_ln_qkv<<<N, 256, 0, stream>>>(node_features, ln_w, ln_b, bq, bk, bv,
                                    wfrag, qg, kg, vg);
    k_edge_attn<<<E, 256, 0, stream>>>(qg, kg, vg, node_masks, etw,
                                       eidx, etyp, ctx, E);
    k_out<<<N, 256, 0, stream>>>(ctx, wfrag, bo, node_features, node_masks, out);
}
